// GroupedLinear_57208964383021
// MI455X (gfx1250) — compile-verified, matrix operand order checked by reading, not run on hardware
//
#include <hip/hip_runtime.h>

// ---- problem constants (match reference) ----
#define Bdim 8192
#define Cdim 4096
#define Gdim 16
#define Sdim 256
#define Rdim 512
#define OUTN (Gdim * Rdim)   // 8192

// ---- tiling ----
#define BM 128               // rows per workgroup (A block persisted in LDS)
#define NCHUNK 64            // N processed per pass (all of R covered in-kernel)
#define NCHUNKS (Rdim / NCHUNK)  // 8
#define BK 32                // K per WMMA stage
#define KSTAGES (Sdim / BK)      // 8
#define A_LDK (Sdim + 8)     // 264 bf16 = 528B row stride (16B-aligned, bank-spread)
#define B_LDK (BK + 8)       // 40 bf16 = 80B row stride

typedef __attribute__((ext_vector_type(16))) __bf16 v16bf;
typedef __attribute__((ext_vector_type(8)))  float  v8f;

union FragCast { v16bf v; uint4 q[2]; };
union PairCast { __bf16 h[2]; unsigned int u; };

// dynamic LDS layout (bytes)
#define A_ELEMS   (BM * A_LDK)            // per hi/lo plane
#define B_ELEMS   (NCHUNK * B_LDK)        // per buffer per hi/lo plane
#define SMEM_BYTES ((size_t)(2 * A_ELEMS + 4 * B_ELEMS) * 2 + Sdim * 4)  // 156672

__global__ __launch_bounds__(256)
void grouped_linear_wmma_persistA(const float* __restrict__ x,
                                  const float* __restrict__ W,
                                  const float* __restrict__ bias,
                                  const int*   __restrict__ perm,
                                  float* __restrict__ out) {
  extern __shared__ __align__(16) char smem[];
  __bf16* As_hi = (__bf16*)smem;                 // [BM][A_LDK]
  __bf16* As_lo = As_hi + A_ELEMS;               // [BM][A_LDK]
  __bf16* Bs_hi = As_lo + A_ELEMS;               // [2][NCHUNK][B_LDK]
  __bf16* Bs_lo = Bs_hi + 2 * B_ELEMS;           // [2][NCHUNK][B_LDK]
  int*    permS = (int*)(Bs_lo + 2 * B_ELEMS);   // [Sdim]

  const int tid = threadIdx.x;
  const int g   = blockIdx.x;        // fast grid dim: co-schedule all groups per row band
  const int m0  = blockIdx.y * BM;

  permS[tid] = perm[g * Sdim + tid];
  __syncthreads();

  // ================= Phase 1: gather full A block once, split bf16 hi/lo ========
  // Thread owns a PAIR of adjacent gathered columns and half the rows, so hi/lo
  // pairs pack into b32 LDS stores (matches v_cvt_pk_bf16_f32 codegen).
  {
    const int cpair = tid & 127;              // column pair index: cols 2c, 2c+1
    const int rhalf = (tid >> 7) * 64;        // rows [rhalf, rhalf+64)
    const int col0  = permS[2 * cpair];
    const int col1  = permS[2 * cpair + 1];
    const float* xp0 = x + (size_t)(m0 + rhalf) * Cdim + col0;
    const float* xp1 = x + (size_t)(m0 + rhalf) * Cdim + col1;
    unsigned int* AhP = (unsigned int*)As_hi;  // A_LDK is even: 132 dwords/row
    unsigned int* AlP = (unsigned int*)As_lo;
#pragma unroll
    for (int c = 0; c < 8; ++c) {
      float a0[8], a1[8];
#pragma unroll
      for (int j = 0; j < 8; ++j) {           // clause of 16 scattered b32 loads
        a0[j] = xp0[(size_t)(c * 8 + j) * Cdim];
        a1[j] = xp1[(size_t)(c * 8 + j) * Cdim];
      }
#pragma unroll
      for (int j = 0; j < 8; ++j) {
        const int row = rhalf + c * 8 + j;
        PairCast hp, lp;
        hp.h[0] = (__bf16)a0[j];
        hp.h[1] = (__bf16)a1[j];
        lp.h[0] = (__bf16)(a0[j] - (float)hp.h[0]);
        lp.h[1] = (__bf16)(a1[j] - (float)hp.h[1]);
        AhP[row * (A_LDK / 2) + cpair] = hp.u;   // packed b32 store
        AlP[row * (A_LDK / 2) + cpair] = lp.u;
      }
    }
  }

  // ================= Phase 2 mappings ==========================================
  // W tile staging: 8 contiguous floats per thread
  const int b_tn = tid >> 2;         // 0..63 : W row within chunk
  const int b_tk = (tid & 3) * 8;    // 0,8,16,24
  const float* Wg = W + (size_t)g * Rdim * Sdim;
  float breg[8];

  auto gloadB = [&](int nc, int ks) {
    const float* wp = Wg + (size_t)(nc * NCHUNK + b_tn) * Sdim + ks * BK + b_tk;
    const float4 q0 = *(const float4*)(wp);
    const float4 q1 = *(const float4*)(wp + 4);
    breg[0] = q0.x; breg[1] = q0.y; breg[2] = q0.z; breg[3] = q0.w;
    breg[4] = q1.x; breg[5] = q1.y; breg[6] = q1.z; breg[7] = q1.w;
  };

  auto storeB = [&](int buf) {
    __bf16* bh = Bs_hi + buf * B_ELEMS + b_tn * B_LDK + b_tk;
    __bf16* bl = Bs_lo + buf * B_ELEMS + b_tn * B_LDK + b_tk;
#pragma unroll
    for (int j = 0; j < 8; ++j) {
      const float v = breg[j];
      const __bf16 h = (__bf16)v;
      const __bf16 l = (__bf16)(v - (float)h);
      bh[j] = h;
      bl[j] = l;
    }
  };

  // wave / lane mapping: 8 waves = 4 (M) x 2 (N), each 32x32 = 2x2 WMMA tiles
  const int lane   = tid & 31;
  const int wid    = tid >> 5;
  const int wmBase = (wid & 3) * 32;
  const int wnBase = (wid >> 2) * 32;
  const int lr     = lane & 15;
  const int khalf  = (lane >> 4) * 8;   // documented 16-bit operand K-half split

  auto loadFragA = [&](const __bf16* base, int rbase, int k0) -> v16bf {
    FragCast f;
    const __bf16* p = base + (size_t)(rbase + lr) * A_LDK + k0 + khalf;
    f.q[0] = *(const uint4*)(p);
    f.q[1] = *(const uint4*)(p + 16);
    return f.v;
  };
  auto loadFragB = [&](const __bf16* bufbase, int rbase) -> v16bf {
    FragCast f;
    const __bf16* p = bufbase + (size_t)(rbase + lr) * B_LDK + khalf;
    f.q[0] = *(const uint4*)(p);
    f.q[1] = *(const uint4*)(p + 16);
    return f.v;
  };

  __syncthreads();   // A block visible to all waves

  // ================= Phase 2: loop over N chunks, K pipelined through LDS =======
  for (int nc = 0; nc < NCHUNKS; ++nc) {
    v8f acc[2][2];
#pragma unroll
    for (int mi = 0; mi < 2; ++mi)
#pragma unroll
      for (int ni = 0; ni < 2; ++ni)
#pragma unroll
        for (int i = 0; i < 8; ++i)
          acc[mi][ni][i] = 0.0f;

    gloadB(nc, 0);
    storeB(0);
    __syncthreads();

    for (int ks = 0; ks < KSTAGES; ++ks) {
      if (ks + 1 < KSTAGES) gloadB(nc, ks + 1);   // overlap next W tile with WMMAs

      const int k0 = ks * BK;
      const __bf16* bhbuf = Bs_hi + (ks & 1) * B_ELEMS;
      const __bf16* blbuf = Bs_lo + (ks & 1) * B_ELEMS;

      v16bf ah[2], al[2], bh[2], bl[2];
      ah[0] = loadFragA(As_hi, wmBase,      k0);
      ah[1] = loadFragA(As_hi, wmBase + 16, k0);
      al[0] = loadFragA(As_lo, wmBase,      k0);
      al[1] = loadFragA(As_lo, wmBase + 16, k0);
      bh[0] = loadFragB(bhbuf, wnBase);
      bh[1] = loadFragB(bhbuf, wnBase + 16);
      bl[0] = loadFragB(blbuf, wnBase);
      bl[1] = loadFragB(blbuf, wnBase + 16);

#pragma unroll
      for (int mi = 0; mi < 2; ++mi)
#pragma unroll
        for (int ni = 0; ni < 2; ++ni) {
          // bf16x3 fp32 emulation: small terms first, hi*hi last
          acc[mi][ni] = __builtin_amdgcn_wmma_f32_16x16x32_bf16(
              false, al[mi], false, bh[ni], (short)0, acc[mi][ni], false, false);
          acc[mi][ni] = __builtin_amdgcn_wmma_f32_16x16x32_bf16(
              false, ah[mi], false, bl[ni], (short)0, acc[mi][ni], false, false);
          acc[mi][ni] = __builtin_amdgcn_wmma_f32_16x16x32_bf16(
              false, ah[mi], false, bh[ni], (short)0, acc[mi][ni], false, false);
        }

      if (ks + 1 < KSTAGES) storeB((ks + 1) & 1);  // write other buffer, then barrier
      __syncthreads();
    }

    // ---- epilogue for this N chunk: bias add + non-temporal fp32 stores ----
    const int n0 = nc * NCHUNK;
#pragma unroll
    for (int mi = 0; mi < 2; ++mi)
#pragma unroll
      for (int ni = 0; ni < 2; ++ni) {
        const int col  = wnBase + ni * 16 + lr;
        const int row0 = wmBase + mi * 16 + (lane >> 4) * 8;
        const float bv = bias[g * Rdim + n0 + col];
        float* op = out + (size_t)(m0 + row0) * OUTN + (size_t)g * Rdim + n0 + col;
#pragma unroll
        for (int i = 0; i < 8; ++i)
          __builtin_nontemporal_store(acc[mi][ni][i] + bv, op + (size_t)i * OUTN);
      }
    // no extra barrier needed: next chunk's first storeB targets buffer 0, whose
    // last readers finished before the ks=6 barrier of this chunk
  }
}

extern "C" void kernel_launch(void* const* d_in, const int* in_sizes, int n_in,
                              void* d_out, int out_size, void* d_ws, size_t ws_size,
                              hipStream_t stream) {
  const float* x    = (const float*)d_in[0];
  const float* W    = (const float*)d_in[1];
  const float* bias = (const float*)d_in[2];
  const int*   perm = (const int*)d_in[3];
  float* out = (float*)d_out;

  dim3 grid(Gdim, Bdim / BM);   // (16, 64): g varies fastest -> L2 consumes whole x lines
  grouped_linear_wmma_persistA<<<grid, dim3(256), SMEM_BYTES, stream>>>(x, W, bias, perm, out);
}
